// AttentionWithRelativePositioning_50964081935049
// MI455X (gfx1250) — matrix operand order repaired and checked by reading, not compiled, not run on hardware
//
#include <hip/hip_runtime.h>
#include <stdint.h>

typedef unsigned short u16;
typedef unsigned int   u32;

typedef __attribute__((ext_vector_type(16))) __bf16 v16bf;
typedef __attribute__((ext_vector_type(8)))  float  v8f;

#define DEV __device__ __forceinline__

constexpr int Bn  = 4;
constexpr int Sq  = 1024;
constexpr int Dm  = 1024;
constexpr int Hh  = 16;
constexpr int DKh = 64;
constexpr int NB  = 65;   // 2*P+1 relative buckets

// ---------------- bf16 conversion helpers (HW packed cvt when available) ----
DEV u16 f2bf_sw(float f) {         // software RNE fallback
  union { float f; u32 u; } v; v.f = f;
  u32 r = v.u + 0x7fffu + ((v.u >> 16) & 1u);
  return (u16)(r >> 16);
}

#if defined(__AMDGCN__) && __has_builtin(__builtin_amdgcn_cvt_pk_bf16_f32)
#define HAVE_PK_BF16 1
#endif

DEV u32 f2bf2(float lo, float hi) {  // two f32 -> packed bf16 pair (u32)
#ifdef HAVE_PK_BF16
  typedef __attribute__((ext_vector_type(2))) __bf16 bf16x2;
  union { bf16x2 v; u32 u; } t;
  t.v = __builtin_amdgcn_cvt_pk_bf16_f32(lo, hi);
  return t.u;
#else
  return (u32)f2bf_sw(lo) | ((u32)f2bf_sw(hi) << 16);
#endif
}

DEV u16 f2bf1(float f) {
#ifdef HAVE_PK_BF16
  return (u16)(f2bf2(f, 0.f) & 0xffffu);
#else
  return f2bf_sw(f);
#endif
}

// ---------------- WMMA ------------------------------------------------------
DEV v8f wmma_bf16(v16bf a, v16bf b, v8f c) {
  // D = A(16x32 bf16) * B(32x16 bf16) + C(16x16 f32)
  return __builtin_amdgcn_wmma_f32_16x16x32_bf16(false, a, false, b, (short)0, c,
                                                 false, false);
}

// ISA 7.12.2: 16-bit A matrix 16x32. Lane L (L<16) holds K={0..7,16..23},
// lane L+16 holds K={8..15,24..31}; VGPR v holds one K pair.
DEV int kOf(int e, int half) {
  int p = e >> 1, b = e & 1;
  int base = (p < 4) ? (p << 1) : (16 + ((p - 4) << 1));
  return base + b + (half ? 8 : 0);
}

union AFrag { v16bf v; u16 u[16]; u32 w[8]; };

// ---------------- Tensor Data Mover (gfx1250 TDM) ---------------------------
#if defined(__AMDGCN__) && __has_builtin(__builtin_amdgcn_tensor_load_to_lds) && \
    __has_builtin(__builtin_amdgcn_s_wait_tensorcnt)
#define HAVE_TDM 1
typedef __attribute__((ext_vector_type(4))) unsigned int u32x4;
typedef __attribute__((ext_vector_type(8))) int          i32x8;
typedef __attribute__((ext_vector_type(4))) int          i32x4;

// Async-load a tile_y(rows) x tile_x(cols) bf16 tile with row stride
// `stride_elems` from global `gptr` into LDS at byte offset `lds_addr`.
// LDS padding: after every (1<<pad_interval) 8-byte units, insert
// (pad_amount+1) DWORDs -> bank-conflict-free pitches.
DEV void tdm_load_2d_bf16(u32 lds_addr, const void* gptr, u32 tile_x, u32 tile_y,
                          u32 stride_elems, u32 pad_interval, u32 pad_amount) {
  unsigned long long ga = (unsigned long long)(uintptr_t)gptr;
  u32x4 g0;
  g0[0] = 1u;                                               // count=1, user D#
  g0[1] = lds_addr;                                         // lds_addr[31:0]
  g0[2] = (u32)ga;                                          // global_addr[31:0]
  g0[3] = ((u32)(ga >> 32) & 0x01ffffffu) | (2u << 30);     // addr[56:32], type=2
  i32x8 g1;
  g1[0] = (int)((1u << 16)                // data_size = 2 bytes
              | (1u << 20)                // pad_enable
              | (pad_interval << 22)
              | (pad_amount << 25));
  g1[1] = 0;                              // tensor_dim0 = 1<<20 (lo bits 0)
  g1[2] = 16;                             // dim0 hi = 16 ; dim1 lo = 0
  g1[3] = (int)((tile_x << 16) | 16u);    // tile_dim0 ; dim1 hi (dim1 = 1<<20)
  g1[4] = (int)tile_y;                    // tile_dim1
  g1[5] = (int)stride_elems;              // tensor_dim0_stride
  g1[6] = 0;
  g1[7] = 0;
  i32x4 zz4 = {0, 0, 0, 0};
  i32x8 zz8 = {0, 0, 0, 0, 0, 0, 0, 0};
  // clang-23 / therock-10.0 headers: 6-arg form (g0, g1, g2, g3, extra, cpol)
  __builtin_amdgcn_tensor_load_to_lds(g0, g1, zz4, zz4, zz8, 0);
}
#endif

// ---------------------------------------------------------------------------
// Tiled bf16 WMMA GEMM:  C[M,N] = A[M,K] * W[N,K]^T   (M=4096, N=K=1024)
// AMODE: 0 = A is fp32 (convert while staging), 1 = A is bf16 raw
// OMODE: 0 = bf16 head-split [B,H,S,dk], 1 = fp32 [M,N],
//        2 = bf16 head-split transposed [B,H,dk,S]
// Block = 128 threads (4 waves); block tile 64x64, wave tile 16x64.
// ---------------------------------------------------------------------------
template <int AMODE, int OMODE>
__global__ __launch_bounds__(128) void gemm64(const void*  __restrict__ Ap,
                                              const float* __restrict__ Wp,
                                              void*        __restrict__ Outp) {
  __shared__ u16 Xt[64][40];   // [m][k]   64 rows x 32 k (bf16), padded pitch
  __shared__ u16 Wt[32][72];   // [k][n]   transposed weight tile

  const int tid  = threadIdx.x;
  const int lane = tid & 31, wave = tid >> 5;
  const int half = lane >> 4, l16 = lane & 15;
  const int m0 = blockIdx.x * 64, n0 = blockIdx.y * 64;

  const float* Af = (const float*)Ap;
  const u16*   Ab = (const u16*)Ap;

  const v8f z = {0.f, 0.f, 0.f, 0.f, 0.f, 0.f, 0.f, 0.f};
  v8f acc[4];
  for (int t = 0; t < 4; ++t) acc[t] = z;

  for (int k0 = 0; k0 < Dm; k0 += 32) {
    __syncthreads();
    for (int idx = tid; idx < 64 * 16; idx += 128) {   // stage A tile (pairs)
      int r = idx >> 4, kp = (idx & 15) << 1;
      u32 pk;
      if constexpr (AMODE == 0) {
        const float* src = Af + (size_t)(m0 + r) * Dm + k0 + kp;
        pk = f2bf2(src[0], src[1]);
      } else {
        pk = *(const u32*)(Ab + (size_t)(m0 + r) * Dm + k0 + kp);
      }
      *(u32*)&Xt[r][kp] = pk;
    }
    for (int idx = tid; idx < 64 * 16; idx += 128) {   // stage W transposed
      int n = idx >> 4, kp = (idx & 15) << 1;
      const float* src = Wp + (size_t)(n0 + n) * Dm + k0 + kp;
      u32 pk = f2bf2(src[0], src[1]);
      Wt[kp][n]     = (u16)(pk & 0xffffu);
      Wt[kp + 1][n] = (u16)(pk >> 16);
    }
    __syncthreads();

    AFrag a;   // A fragment: 16 rows of this wave
    {
      const u16* xr = &Xt[wave * 16 + l16][0];
      for (int p = 0; p < 8; ++p) {
        int k = kOf(2 * p, half);
        a.w[p] = *(const u32*)(xr + k);
      }
    }
    for (int t = 0; t < 4; ++t) {   // 4 N-subtiles of 16
      AFrag b;                       // B: lane = N col, elements = K
      for (int p = 0; p < 8; ++p) {
        int k = kOf(2 * p, half);
        b.u[2 * p]     = Wt[k][t * 16 + l16];
        b.u[2 * p + 1] = Wt[k + 1][t * 16 + l16];
      }
      acc[t] = wmma_bf16(a.v, b.v, acc[t]);
    }
  }

  // epilogue; C layout: M = v + 8*half, N = lane&15
  for (int t = 0; t < 4; ++t) {
    for (int v = 0; v < 8; ++v) {
      int M   = v + 8 * half;
      int row = m0 + wave * 16 + M;
      int col = n0 + t * 16 + l16;
      float val = acc[t][v];
      if constexpr (OMODE == 0) {
        int b = row >> 10, s = row & 1023;
        int h = col >> 6,  d = col & 63;
        ((u16*)Outp)[(((size_t)(b * Hh + h)) * Sq + s) * DKh + d] = f2bf1(val);
      } else if constexpr (OMODE == 2) {
        int b = row >> 10, s = row & 1023;
        int h = col >> 6,  d = col & 63;
        ((u16*)Outp)[(((size_t)(b * Hh + h)) * DKh + d) * Sq + s] = f2bf1(val);
      } else {
        ((float*)Outp)[(size_t)row * Dm + col] = val;
      }
    }
  }
}

// ---------------------------------------------------------------------------
// Fused flash attention + relative-position bias.
// Block = 128 threads (4 waves); each wave owns a 16-row i-tile; block covers
// 64 rows of one (b,h). Two-pass online softmax (pass1: row max, pass2:
// exp / l / O via WMMA / bucket weights W via LDS atomics). K (d-major) and
// V (j-major) chunks are staged with the Tensor Data Mover when available;
// B fragments are gathered from LDS with 16-bit reads, lane = column.
// ---------------------------------------------------------------------------
__global__ __launch_bounds__(128) void flash_relpos(const u16*  __restrict__ Qh,
                                                    const u16*  __restrict__ KhT,
                                                    const u16*  __restrict__ Vh,
                                                    const float* __restrict__ ek,
                                                    const float* __restrict__ ev,
                                                    u16*  __restrict__ Ows) {
  __shared__ u16   KtT[64][36];       // K chunk [d][j] bf16 (TDM-padded pitch)
  __shared__ u16   Vt[32][72];        // V chunk [j][d] bf16
  __shared__ u16   EkT[64][84];       // emb_k transposed [d][r], r 65..79 zero
  __shared__ float Tl[4][16][68];     // per-wave bias table T[i,r]
  __shared__ float Wl[4][16][68];     // per-wave bucket weights W[i,r]
  __shared__ u16   Ps[4][16][34];     // per-wave P tile (C-layout -> A-layout)

  const int tid  = threadIdx.x;
  const int lane = tid & 31, wave = tid >> 5;
  const int half = lane >> 4, l16 = lane & 15;

  const int bh    = blockIdx.x >> 4;  // b*H + h
  const int itile = blockIdx.x & 15;
  const int i0    = itile * 64 + wave * 16;

  const u16* Qb  = Qh  + (size_t)bh * Sq * DKh;
  const u16* KbT = KhT + (size_t)bh * DKh * Sq;   // [d][s]
  const u16* Vb  = Vh  + (size_t)bh * Sq * DKh;   // [s][d]

  const v8f z = {0.f, 0.f, 0.f, 0.f, 0.f, 0.f, 0.f, 0.f};

  // Q A-fragments for dk=64 (two K=32 fragments), persistent in VGPRs
  AFrag qa[2];
  {
    const u16* qr = Qb + (size_t)(i0 + l16) * DKh;
    for (int p = 0; p < 8; ++p) {
      int k = kOf(2 * p, half);
      qa[0].w[p] = *(const u32*)(qr + k);
      qa[1].w[p] = *(const u32*)(qr + 32 + k);
    }
  }

  // zero this wave's bucket accumulator
  for (int idx = lane; idx < 16 * 68; idx += 32) Wl[wave][idx / 68][idx % 68] = 0.f;

  // stage emb_k transposed as bf16 in LDS once per block (pair converts)
  for (int idx = tid; idx < 64 * 40; idx += 128) {
    int d = idx / 40, rp = (idx % 40) << 1;
    float lo = (rp     < NB) ? ek[(size_t)rp * DKh + d]       : 0.f;
    float hi = (rp + 1 < NB) ? ek[(size_t)(rp + 1) * DKh + d] : 0.f;
    *(u32*)&EkT[d][rp] = f2bf2(lo, hi);
  }
  __syncthreads();

  // T[i,r] = Q[i] . emb_k[r]  via WMMA (N padded 65 -> 80)
  for (int r0t = 0; r0t < 5; ++r0t) {
    int r0 = r0t * 16;
    v8f c = z;
    for (int kk = 0; kk < 2; ++kk) {
      AFrag b;                        // B: lane = r col, elements = d
      for (int p = 0; p < 8; ++p) {
        int k = kOf(2 * p, half);
        b.u[2 * p]     = EkT[kk * 32 + k][r0 + l16];
        b.u[2 * p + 1] = EkT[kk * 32 + k + 1][r0 + l16];
      }
      c = wmma_bf16(qa[kk].v, b.v, c);
    }
    for (int v = 0; v < 8; ++v) {
      int M = v + 8 * half;
      int r = r0 + l16;
      if (r < NB) Tl[wave][M][r] = c[v];
    }
  }
  asm volatile("s_wait_dscnt 0" ::: "memory");

  const float SCALE = 0.125f;   // 1/sqrt(64)

  // ---------------- pass 1: row maxima -------------------------------------
  float mrow[8];
  for (int v = 0; v < 8; ++v) mrow[v] = -3.0e38f;

  for (int j0 = 0; j0 < Sq; j0 += 32) {
    __syncthreads();
#ifdef HAVE_TDM
    if (wave == 0) {
      // 64 rows (d) x 32 cols (j), row stride S; 8B pad per 64B row -> pitch 72B
      tdm_load_2d_bf16((u32)(uintptr_t)&KtT[0][0], KbT + j0,
                       32, 64, Sq, 3, 1);
      __builtin_amdgcn_s_wait_tensorcnt(0);
    }
#else
    for (int idx = tid; idx < 64 * 16; idx += 128) {
      int d = idx >> 4, jp = (idx & 15) << 1;
      *(u32*)&KtT[d][jp] = *(const u32*)(KbT + (size_t)d * Sq + j0 + jp);
    }
#endif
    __syncthreads();
    for (int nh = 0; nh < 2; ++nh) {
      v8f c = z;
      for (int kk = 0; kk < 2; ++kk) {
        AFrag b;                      // B: lane = j col, elements = d
        for (int p = 0; p < 8; ++p) {
          int k = kOf(2 * p, half);
          b.u[2 * p]     = KtT[kk * 32 + k][nh * 16 + l16];
          b.u[2 * p + 1] = KtT[kk * 32 + k + 1][nh * 16 + l16];
        }
        c = wmma_bf16(qa[kk].v, b.v, c);
      }
      for (int v = 0; v < 8; ++v) {
        int M = v + 8 * half;
        int i = i0 + M;
        int j = j0 + nh * 16 + l16;
        int dlt = j - i; dlt = dlt < -32 ? -32 : (dlt > 32 ? 32 : dlt);
        float s = (c[v] + Tl[wave][M][dlt + 32]) * SCALE;
        mrow[v] = fmaxf(mrow[v], s);
      }
    }
  }
  for (int v = 0; v < 8; ++v) {       // reduce across the 16 lanes of a half
    float x = mrow[v];
    x = fmaxf(x, __shfl_xor(x, 1, 32));
    x = fmaxf(x, __shfl_xor(x, 2, 32));
    x = fmaxf(x, __shfl_xor(x, 4, 32));
    x = fmaxf(x, __shfl_xor(x, 8, 32));
    mrow[v] = x;
  }

  // ---------------- pass 2: exp, l, O (WMMA), bucket weights ---------------
  float lrow[8];
  for (int v = 0; v < 8; ++v) lrow[v] = 0.f;
  v8f of[4];
  for (int t = 0; t < 4; ++t) of[t] = z;

  for (int j0 = 0; j0 < Sq; j0 += 32) {
    __syncthreads();
#ifdef HAVE_TDM
    if (wave == 0) {
      tdm_load_2d_bf16((u32)(uintptr_t)&KtT[0][0], KbT + j0,
                       32, 64, Sq, 3, 1);
      // 32 rows (j) x 64 cols (d), row stride 64; 16B pad per 128B -> pitch 144B
      tdm_load_2d_bf16((u32)(uintptr_t)&Vt[0][0], Vb + (size_t)j0 * DKh,
                       64, 32, 64, 4, 3);
      __builtin_amdgcn_s_wait_tensorcnt(0);
    }
#else
    for (int idx = tid; idx < 64 * 16; idx += 128) {
      int d = idx >> 4, jp = (idx & 15) << 1;
      *(u32*)&KtT[d][jp] = *(const u32*)(KbT + (size_t)d * Sq + j0 + jp);
    }
    for (int idx = tid; idx < 32 * 32; idx += 128) {
      int jj = idx >> 5, dp = (idx & 31) << 1;
      *(u32*)&Vt[jj][dp] = *(const u32*)(Vb + (size_t)(j0 + jj) * DKh + dp);
    }
#endif
    __syncthreads();

    for (int nh = 0; nh < 2; ++nh) {
      v8f c = z;
      for (int kk = 0; kk < 2; ++kk) {
        AFrag b;
        for (int p = 0; p < 8; ++p) {
          int k = kOf(2 * p, half);
          b.u[2 * p]     = KtT[kk * 32 + k][nh * 16 + l16];
          b.u[2 * p + 1] = KtT[kk * 32 + k + 1][nh * 16 + l16];
        }
        c = wmma_bf16(qa[kk].v, b.v, c);
      }
      for (int v = 0; v < 8; ++v) {
        int M = v + 8 * half;
        int i = i0 + M;
        int j = j0 + nh * 16 + l16;
        int dlt = j - i;
        int r = (dlt < -32 ? -32 : (dlt > 32 ? 32 : dlt)) + 32;
        float s = (c[v] + Tl[wave][M][r]) * SCALE;
        float p = __expf(s - mrow[v]);
        lrow[v] += p;
        atomicAdd(&Wl[wave][M][r], p);          // ds_add_f32
        Ps[wave][M][nh * 16 + l16] = f2bf1(p);  // stage P for A-layout gather
      }
    }
    asm volatile("s_wait_dscnt 0" ::: "memory");

    AFrag pa;   // P as A fragment (16x32)
    {
      const u16* pr = &Ps[wave][l16][0];
      for (int p = 0; p < 8; ++p) {
        int k = kOf(2 * p, half);
        pa.w[p] = *(const u32*)(pr + k);
      }
    }
    for (int t = 0; t < 4; ++t) {     // O += P * V  (4 d-subtiles of 16)
      AFrag b;                        // B: lane = d col, elements = j
      for (int p = 0; p < 8; ++p) {
        int k = kOf(2 * p, half);
        b.u[2 * p]     = Vt[k][t * 16 + l16];
        b.u[2 * p + 1] = Vt[k + 1][t * 16 + l16];
      }
      of[t] = wmma_bf16(pa.v, b.v, of[t]);
    }
  }

  for (int v = 0; v < 8; ++v) {       // final l reduction -> 1/l
    float x = lrow[v];
    x += __shfl_xor(x, 1, 32);
    x += __shfl_xor(x, 2, 32);
    x += __shfl_xor(x, 4, 32);
    x += __shfl_xor(x, 8, 32);
    lrow[v] = 1.f / x;
  }
  asm volatile("s_wait_dscnt 0" ::: "memory");

  // epilogue: out = (O + W @ emb_v) / l, written bf16 as [B,S,(h,dk)]
  const int b = bh >> 4, h = bh & 15;
  for (int t = 0; t < 4; ++t) {
    float rel[8];
    for (int v = 0; v < 8; ++v) rel[v] = 0.f;
    int d = t * 16 + l16;
    for (int r = 0; r < NB; ++r) {
      float evv = ev[(size_t)r * DKh + d];
      for (int v = 0; v < 8; ++v) {
        int M = v + 8 * half;
        rel[v] += Wl[wave][M][r] * evv;
      }
    }
    for (int v = 0; v < 8; ++v) {
      int M = v + 8 * half;
      int i = i0 + M;
      float val = (of[t][v] + rel[v]) * lrow[v];
      Ows[((size_t)(b * Sq + i)) * Dm + h * DKh + d] = f2bf1(val);
    }
  }
}

// ---------------------------------------------------------------------------
// Residual + LayerNorm: out = LN(query + Y) * gamma + beta, one block per row
// ---------------------------------------------------------------------------
__global__ __launch_bounds__(256) void ln_residual(const float* __restrict__ Yp,
                                                   const float* __restrict__ Xq,
                                                   const float* __restrict__ gamma,
                                                   const float* __restrict__ beta,
                                                   float* __restrict__ out) {
  const int row = blockIdx.x;
  const int tid = threadIdx.x;
  __shared__ float rs[8], rs2[8];

  float x[4], s = 0.f, s2 = 0.f;
  for (int q = 0; q < 4; ++q) {
    int c = tid + q * 256;
    float v = Xq[(size_t)row * Dm + c] + Yp[(size_t)row * Dm + c];
    x[q] = v; s += v; s2 += v * v;
  }
  for (int m = 1; m < 32; m <<= 1) {
    s  += __shfl_xor(s,  m, 32);
    s2 += __shfl_xor(s2, m, 32);
  }
  if ((tid & 31) == 0) { rs[tid >> 5] = s; rs2[tid >> 5] = s2; }
  __syncthreads();
  float ts = 0.f, ts2 = 0.f;
  for (int i = 0; i < 8; ++i) { ts += rs[i]; ts2 += rs2[i]; }
  float mean = ts  * (1.f / 1024.f);
  float var  = ts2 * (1.f / 1024.f) - mean * mean;
  float rstd = rsqrtf(var + 1e-5f);
  for (int q = 0; q < 4; ++q) {
    int c = tid + q * 256;
    out[(size_t)row * Dm + c] = (x[q] - mean) * rstd * gamma[c] + beta[c];
  }
}

// ---------------------------------------------------------------------------
extern "C" void kernel_launch(void* const* d_in, const int* in_sizes, int n_in,
                              void* d_out, int out_size, void* d_ws, size_t ws_size,
                              hipStream_t stream) {
  (void)in_sizes; (void)n_in; (void)out_size; (void)ws_size;

  const float* query = (const float*)d_in[0];
  const float* key_  = (const float*)d_in[1];
  const float* value = (const float*)d_in[2];
  const float* w_q   = (const float*)d_in[3];
  const float* w_k   = (const float*)d_in[4];
  const float* w_v   = (const float*)d_in[5];
  const float* w_o   = (const float*)d_in[6];
  const float* emb_k = (const float*)d_in[7];
  const float* emb_v = (const float*)d_in[8];
  const float* ln_g  = (const float*)d_in[9];
  const float* ln_b  = (const float*)d_in[10];

  // workspace layout (48 MB total)
  char* ws = (char*)d_ws;
  u16*   Qh  = (u16*)(ws + (size_t) 0);                 //  8 MB bf16 [B,H,S,dk]
  u16*   KhT = (u16*)(ws + (size_t) 8 * 1024 * 1024);   //  8 MB bf16 [B,H,dk,S]
  u16*   Vh  = (u16*)(ws + (size_t)16 * 1024 * 1024);   //  8 MB bf16 [B,H,S,dk]
  u16*   Ow  = (u16*)(ws + (size_t)24 * 1024 * 1024);   //  8 MB bf16 [B*S, D]
  float* Y   = (float*)(ws + (size_t)32 * 1024 * 1024); // 16 MB fp32 [B*S, D]

  dim3 g(64, 16), blk(128);
  gemm64<0, 0><<<g, blk, 0, stream>>>((const void*)query, w_q, (void*)Qh);
  gemm64<0, 2><<<g, blk, 0, stream>>>((const void*)key_,  w_k, (void*)KhT);
  gemm64<0, 0><<<g, blk, 0, stream>>>((const void*)value, w_v, (void*)Vh);

  flash_relpos<<<dim3(1024), blk, 0, stream>>>(Qh, KhT, Vh, emb_k, emb_v, Ow);

  gemm64<1, 1><<<g, blk, 0, stream>>>((const void*)Ow, w_o, (void*)Y);

  ln_residual<<<dim3(4096), dim3(256), 0, stream>>>(Y, query, ln_g, ln_b,
                                                    (float*)d_out);
}
